// Mamba_MIL_52896817218169
// MI455X (gfx1250) — compile-verified, matrix operand order checked by reading, not run on hardware
//
#include <hip/hip_runtime.h>
#include <hip/hip_bf16.h>
#include <stdint.h>

// ---------------------------------------------------------------------------
// Mamba-MIL pipeline for gfx1250 (MI455X).
//  - GEMMs: v_wmma_f32_16x16x32_bf16, 32x64 tile/wave, K & LDA templated so
//    all fragment addresses fold into 24-bit immediate offsets on 2 base
//    pointers (no per-iter 64-bit pointer arithmetic, no spills).
//  - Selective scan: 3-phase chunked scan (16 segments x 256 steps).
// ---------------------------------------------------------------------------

#define D_MODEL  512
#define D_STATE  16
#define D_CONV   4
#define D_INNER  1024
#define DT_RANK  32
#define N_LAYERS 2
#define IN_DIM   1024
#define N_CLS    2
#define N_INST   4096

#define SEG_LEN  256
#define N_SEG    (N_INST / SEG_LEN)   // 16

typedef __attribute__((ext_vector_type(16))) __bf16 v16bf;
typedef __attribute__((ext_vector_type(8)))  float  v8f;
typedef unsigned int uint32;

union ABFrag { uint32 u[8]; v16bf v; };

__device__ __forceinline__ uint16_t f2bf(float f) {
    uint32 u = __builtin_bit_cast(uint32, f);
    uint32 r = u + 0x7FFFu + ((u >> 16) & 1u);   // round-to-nearest-even
    return (uint16_t)(r >> 16);
}

__device__ __forceinline__ float siluf(float x) {
    return x / (1.f + __expf(-x));
}

// ---------------------------------------------------------------------------
// f32 -> bf16 conversion
// ---------------------------------------------------------------------------
__global__ void cvt_bf16_kernel(const float* __restrict__ src,
                                uint16_t* __restrict__ dst, int n) {
    int i = blockIdx.x * 256 + threadIdx.x;
    if (i < n) dst[i] = f2bf(src[i]);
}

// ---------------------------------------------------------------------------
// WMMA bf16 GEMM:  out[M,N] = act( A[M,K] @ B[N,K]^T + bias ) (+ resid)
//  - K (reduction dim), LDA (A row stride), ACT compile-time:
//      ACT: 0=none 1=relu 2=tanh 3=softplus
//  - B packed (N x K); out row stride N
// Each wave computes a 32x64 tile (2 M-frags x 4 N-frags, 8 accumulators).
// Block = 128 threads (4 waves) covering 128 rows; grid = (N/64, M/128).
// All intra-tile displacements (16*LDA, t*16*K, k) are compile-time constants
// and fold into global_load immediate offsets (24-bit signed on gfx1250).
// ---------------------------------------------------------------------------
template <int K, int LDA, int ACT>
__global__ __launch_bounds__(128, 1)
void gemm_wmma_bf16(const uint16_t* __restrict__ A,
                    const uint16_t* __restrict__ B,
                    const float* __restrict__ bias,
                    const float* __restrict__ resid,
                    float* __restrict__ outf,
                    uint16_t* __restrict__ outb,
                    int M, int N) {
    const int lane = threadIdx.x & 31;
    const int wv   = threadIdx.x >> 5;
    const int m0   = blockIdx.y * 128 + wv * 32;
    const int n0   = blockIdx.x * 64;

    const int hl = lane >> 4;        // 16-lane half selector
    const int r  = lane & 15;        // row within fragment

    // Two base pointers; everything else is an immediate offset.
    const uint16_t* aptr = A + (size_t)(m0 + r) * LDA + hl * 8;
    const uint16_t* bptr = B + (size_t)(n0 + r) * K + hl * 8;

    v8f acc[2][4];
#pragma unroll
    for (int i = 0; i < 2; ++i)
#pragma unroll
        for (int t = 0; t < 4; ++t)
            acc[i][t] = (v8f){0.f, 0.f, 0.f, 0.f, 0.f, 0.f, 0.f, 0.f};

#pragma unroll 2
    for (int k = 0; k < K; k += 32) {
        ABFrag a[2], b[4];
#pragma unroll
        for (int p = 0; p < 8; ++p) {
            const int off = k + 2 * p + ((p >= 4) ? 8 : 0);
            a[0].u[p] = *(const uint32*)(aptr + off);
            a[1].u[p] = *(const uint32*)(aptr + 16 * LDA + off);
#pragma unroll
            for (int t = 0; t < 4; ++t)
                b[t].u[p] = *(const uint32*)(bptr + t * 16 * K + off);
        }
#pragma unroll
        for (int i = 0; i < 2; ++i)
#pragma unroll
            for (int t = 0; t < 4; ++t)
                acc[i][t] = __builtin_amdgcn_wmma_f32_16x16x32_bf16(
                    false, a[i].v, false, b[t].v, (short)0, acc[i][t],
                    false, false);
    }

#pragma unroll
    for (int t = 0; t < 4; ++t) {
        const int n  = n0 + t * 16 + r;   // C/D layout: both halves carry N=r
        const float bv = bias ? bias[n] : 0.f;
#pragma unroll
        for (int i = 0; i < 2; ++i) {
#pragma unroll
            for (int vg = 0; vg < 8; ++vg) {
                const int m = m0 + i * 16 + vg + hl * 8;
                float x = acc[i][t][vg] + bv;
                if (ACT == 1)      x = fmaxf(x, 0.f);
                else if (ACT == 2) x = tanhf(x);
                else if (ACT == 3) x = (x > 20.f) ? x : log1pf(__expf(x));
                if (resid) x += resid[(size_t)m * N + n];
                if (outf)  outf[(size_t)m * N + n] = x;
                if (outb)  outb[(size_t)m * N + n] = f2bf(x);
            }
        }
    }
}

// ---------------------------------------------------------------------------
// LayerNorm: one wave32 per 512-wide row (16 elems/lane)
// ---------------------------------------------------------------------------
__global__ void ln_kernel(const float* __restrict__ in,
                          const float* __restrict__ w,
                          const float* __restrict__ b,
                          float* __restrict__ outf,
                          uint16_t* __restrict__ outb) {
    const int row  = (blockIdx.x * blockDim.x + threadIdx.x) >> 5;
    const int lane = threadIdx.x & 31;
    const float* rp = in + (size_t)row * D_MODEL;

    float v[16];
    float s = 0.f;
#pragma unroll
    for (int i = 0; i < 16; ++i) { v[i] = rp[lane + i * 32]; s += v[i]; }
#pragma unroll
    for (int off = 16; off > 0; off >>= 1) s += __shfl_xor(s, off, 32);
    const float mean = s * (1.f / 512.f);

    float vs = 0.f;
#pragma unroll
    for (int i = 0; i < 16; ++i) { float t = v[i] - mean; vs += t * t; }
#pragma unroll
    for (int off = 16; off > 0; off >>= 1) vs += __shfl_xor(vs, off, 32);
    const float rstd = rsqrtf(vs * (1.f / 512.f) + 1e-5f);

#pragma unroll
    for (int i = 0; i < 16; ++i) {
        const int c = lane + i * 32;
        const float o = (v[i] - mean) * rstd * w[c] + b[c];
        if (outf) outf[(size_t)row * D_MODEL + c] = o;
        if (outb) outb[(size_t)row * D_MODEL + c] = f2bf(o);
    }
}

// ---------------------------------------------------------------------------
// Depthwise causal conv (k=4) over L + bias + SiLU; xi = xz[:, 0:1024]
// ---------------------------------------------------------------------------
__global__ void conv_silu_kernel(const float* __restrict__ xz,
                                 const float* __restrict__ cw,
                                 const float* __restrict__ cb,
                                 float* __restrict__ xc,
                                 uint16_t* __restrict__ xcb) {
    const int idx = blockIdx.x * 256 + threadIdx.x;
    const int d = idx & (D_INNER - 1);
    const int l = idx >> 10;
    float s = cb[d];
#pragma unroll
    for (int k = 0; k < D_CONV; ++k) {
        const int li = l - (D_CONV - 1) + k;
        if (li >= 0) s += cw[d * D_CONV + k] * xz[(size_t)li * (2 * D_INNER) + d];
    }
    const float o = siluf(s);
    xc[idx]  = o;
    xcb[idx] = f2bf(o);
}

// ---------------------------------------------------------------------------
// Chunked selective scan.
// h <- exp(dt*A)*h + B*(dt*xc); over a segment the composed map is
// h_out = exp(A * sum(dt)) * h_in + local.
// ---------------------------------------------------------------------------
__global__ void scan_phase_a(const float* __restrict__ dt,
                             const float* __restrict__ xc,
                             const float* __restrict__ dbc,
                             const float* __restrict__ A_log,
                             float* __restrict__ hseg,
                             float* __restrict__ sumdt) {
    __shared__ float sB[64 * D_STATE];
    const int seg = blockIdx.y;
    const int d   = blockIdx.x * 256 + threadIdx.x;

    float A[D_STATE], h[D_STATE];
#pragma unroll
    for (int n = 0; n < D_STATE; ++n) {
        A[n] = -__expf(A_log[(size_t)d * D_STATE + n]);
        h[n] = 0.f;
    }
    float sdt = 0.f;

    for (int l0 = seg * SEG_LEN; l0 < (seg + 1) * SEG_LEN; l0 += 64) {
        __syncthreads();
        for (int idx = threadIdx.x; idx < 64 * D_STATE; idx += 256) {
            const int s = idx >> 4, n = idx & 15;
            sB[idx] = dbc[(size_t)(l0 + s) * 64 + DT_RANK + n];
        }
        __syncthreads();
#pragma unroll 4
        for (int s = 0; s < 64; ++s) {
            const int l = l0 + s;
            const float dtv = dt[(size_t)l * D_INNER + d];
            const float xv  = xc[(size_t)l * D_INNER + d];
            const float dx  = dtv * xv;
            sdt += dtv;
#pragma unroll
            for (int n = 0; n < D_STATE; ++n)
                h[n] = __expf(dtv * A[n]) * h[n] + sB[s * D_STATE + n] * dx;
        }
    }
#pragma unroll
    for (int n = 0; n < D_STATE; ++n)
        hseg[((size_t)seg * D_INNER + d) * D_STATE + n] = h[n];
    sumdt[seg * D_INNER + d] = sdt;
}

__global__ void scan_phase_b(const float* __restrict__ hseg,
                             const float* __restrict__ sumdt,
                             const float* __restrict__ A_log,
                             float* __restrict__ carry) {
    const int idx = blockIdx.x * 256 + threadIdx.x;   // d*16 + n
    const int d = idx >> 4;
    const float An = -__expf(A_log[idx]);
    float c = 0.f;
#pragma unroll
    for (int s = 0; s < N_SEG; ++s) {
        carry[((size_t)s * D_INNER + d) * D_STATE + (idx & 15)] = c;
        c = __expf(An * sumdt[s * D_INNER + d]) * c
            + hseg[((size_t)s * D_INNER + d) * D_STATE + (idx & 15)];
    }
}

__global__ void scan_phase_c(const float* __restrict__ dt,
                             const float* __restrict__ xc,
                             const float* __restrict__ dbc,
                             const float* __restrict__ xz,
                             const float* __restrict__ A_log,
                             const float* __restrict__ D_p,
                             const float* __restrict__ carry,
                             uint16_t* __restrict__ yb) {
    __shared__ float sB[64 * D_STATE];
    __shared__ float sC[64 * D_STATE];
    const int seg = blockIdx.y;
    const int d   = blockIdx.x * 256 + threadIdx.x;

    float A[D_STATE], h[D_STATE];
#pragma unroll
    for (int n = 0; n < D_STATE; ++n) {
        A[n] = -__expf(A_log[(size_t)d * D_STATE + n]);
        h[n] = carry[((size_t)seg * D_INNER + d) * D_STATE + n];
    }
    const float Dp = D_p[d];

    for (int l0 = seg * SEG_LEN; l0 < (seg + 1) * SEG_LEN; l0 += 64) {
        __syncthreads();
        for (int idx = threadIdx.x; idx < 64 * D_STATE; idx += 256) {
            const int s = idx >> 4, n = idx & 15;
            sB[idx] = dbc[(size_t)(l0 + s) * 64 + DT_RANK + n];
            sC[idx] = dbc[(size_t)(l0 + s) * 64 + DT_RANK + D_STATE + n];
        }
        __syncthreads();
#pragma unroll 4
        for (int s = 0; s < 64; ++s) {
            const int l = l0 + s;
            const float dtv = dt[(size_t)l * D_INNER + d];
            const float xv  = xc[(size_t)l * D_INNER + d];
            const float zv  = xz[(size_t)l * (2 * D_INNER) + D_INNER + d];
            const float dx  = dtv * xv;
            float acc = 0.f;
#pragma unroll
            for (int n = 0; n < D_STATE; ++n) {
                h[n] = __expf(dtv * A[n]) * h[n] + sB[s * D_STATE + n] * dx;
                acc += h[n] * sC[s * D_STATE + n];
            }
            const float y = (acc + Dp * xv) * siluf(zv);
            yb[(size_t)l * D_INNER + d] = f2bf(y);
        }
    }
}

// ---------------------------------------------------------------------------
// a[i] = dot(t[i,0:128], w2) + b2  — one wave per instance
// ---------------------------------------------------------------------------
__global__ void attn_score_kernel(const float* __restrict__ t,
                                  const float* __restrict__ w2,
                                  const float* __restrict__ b2,
                                  float* __restrict__ a) {
    const int row  = (blockIdx.x * blockDim.x + threadIdx.x) >> 5;
    const int lane = threadIdx.x & 31;
    const float* rp = t + (size_t)row * 128;
    float s = 0.f;
#pragma unroll
    for (int i = 0; i < 4; ++i) s += rp[lane + i * 32] * w2[lane + i * 32];
#pragma unroll
    for (int off = 16; off > 0; off >>= 1) s += __shfl_xor(s, off, 32);
    if (lane == 0) a[row] = s + b2[0];
}

// ---------------------------------------------------------------------------
// Softmax over 4096 scores, single block of 1024 threads
// ---------------------------------------------------------------------------
__global__ void softmax_kernel(const float* __restrict__ a,
                               float* __restrict__ w) {
    __shared__ float sm[32];
    const int tid = threadIdx.x;
    float v[4];
    float m = -1e30f;
#pragma unroll
    for (int i = 0; i < 4; ++i) { v[i] = a[tid + i * 1024]; m = fmaxf(m, v[i]); }
#pragma unroll
    for (int off = 16; off > 0; off >>= 1) m = fmaxf(m, __shfl_xor(m, off, 32));
    if ((tid & 31) == 0) sm[tid >> 5] = m;
    __syncthreads();
    if (tid < 32) {
        float t = sm[tid];
#pragma unroll
        for (int off = 16; off > 0; off >>= 1) t = fmaxf(t, __shfl_xor(t, off, 32));
        if (tid == 0) sm[0] = t;
    }
    __syncthreads();
    m = sm[0];
    __syncthreads();

    float s = 0.f;
#pragma unroll
    for (int i = 0; i < 4; ++i) { v[i] = __expf(v[i] - m); s += v[i]; }
#pragma unroll
    for (int off = 16; off > 0; off >>= 1) s += __shfl_xor(s, off, 32);
    if ((tid & 31) == 0) sm[tid >> 5] = s;
    __syncthreads();
    if (tid < 32) {
        float t = sm[tid];
#pragma unroll
        for (int off = 16; off > 0; off >>= 1) t += __shfl_xor(t, off, 32);
        if (tid == 0) sm[0] = t;
    }
    __syncthreads();
    const float inv = 1.f / sm[0];
#pragma unroll
    for (int i = 0; i < 4; ++i) w[tid + i * 1024] = v[i] * inv;
}

// ---------------------------------------------------------------------------
// pooled[c] = sum_i w[i] * h[i,c]
// ---------------------------------------------------------------------------
__global__ void pool_kernel(const float* __restrict__ w,
                            const float* __restrict__ h,
                            float* __restrict__ pooled) {
    const int c = threadIdx.x;
    float s = 0.f;
    for (int i = 0; i < N_INST; ++i) s += w[i] * h[(size_t)i * D_MODEL + c];
    pooled[c] = s;
}

// ---------------------------------------------------------------------------
// logits[k] = dot(pooled, cls_w[k]) + cls_b[k]
// ---------------------------------------------------------------------------
__global__ void logits_kernel(const float* __restrict__ pooled,
                              const float* __restrict__ cw,
                              const float* __restrict__ cb,
                              float* __restrict__ out) {
    const int k    = threadIdx.x >> 5;
    const int lane = threadIdx.x & 31;
    if (k < N_CLS) {
        float s = 0.f;
#pragma unroll
        for (int i = 0; i < 16; ++i)
            s += pooled[lane + i * 32] * cw[k * D_MODEL + lane + i * 32];
#pragma unroll
        for (int off = 16; off > 0; off >>= 1) s += __shfl_xor(s, off, 32);
        if (lane == 0) out[k] = s + cb[k];
    }
}

// ---------------------------------------------------------------------------
// Host orchestration
// ---------------------------------------------------------------------------
static inline dim3 gemm_grid(int M, int N) { return dim3(N / 64, M / 128); }

extern "C" void kernel_launch(void* const* d_in, const int* in_sizes, int n_in,
                              void* d_out, int out_size, void* d_ws, size_t ws_size,
                              hipStream_t stream) {
    const float* x        = (const float*)d_in[0];
    const float* fc1_w    = (const float*)d_in[1];
    const float* fc1_b    = (const float*)d_in[2];
    const float* ln_w     = (const float*)d_in[3];
    const float* ln_b     = (const float*)d_in[4];
    const float* in_proj  = (const float*)d_in[5];
    const float* conv_w   = (const float*)d_in[6];
    const float* conv_b   = (const float*)d_in[7];
    const float* x_proj   = (const float*)d_in[8];
    const float* dt_w     = (const float*)d_in[9];
    const float* dt_b     = (const float*)d_in[10];
    const float* A_log    = (const float*)d_in[11];
    const float* D_p      = (const float*)d_in[12];
    const float* out_proj = (const float*)d_in[13];
    const float* norm_w   = (const float*)d_in[14];
    const float* norm_b   = (const float*)d_in[15];
    const float* attn_w1  = (const float*)d_in[16];
    const float* attn_b1  = (const float*)d_in[17];
    const float* attn_w2  = (const float*)d_in[18];
    const float* attn_b2  = (const float*)d_in[19];
    const float* cls_w    = (const float*)d_in[20];
    const float* cls_b    = (const float*)d_in[21];
    float* out = (float*)d_out;

    char* cur = (char*)d_ws;
    auto alloc = [&](size_t bytes) -> void* {
        void* r = (void*)cur;
        cur += (bytes + 255) & ~(size_t)255;
        return r;
    };

    // bf16 weight copies
    uint16_t* wb_fc1   = (uint16_t*)alloc((size_t)D_MODEL * IN_DIM * 2);
    uint16_t* wb_inprj = (uint16_t*)alloc((size_t)N_LAYERS * 2 * D_INNER * D_MODEL * 2);
    uint16_t* wb_xprj  = (uint16_t*)alloc((size_t)N_LAYERS * 64 * D_INNER * 2);
    uint16_t* wb_dtw   = (uint16_t*)alloc((size_t)N_LAYERS * D_INNER * DT_RANK * 2);
    uint16_t* wb_oprj  = (uint16_t*)alloc((size_t)N_LAYERS * D_MODEL * D_INNER * 2);
    uint16_t* wb_attn1 = (uint16_t*)alloc((size_t)128 * D_MODEL * 2);
    // activations
    uint16_t* xb16   = (uint16_t*)alloc((size_t)N_INST * IN_DIM * 2);
    float*    hbuf   = (float*)   alloc((size_t)N_INST * D_MODEL * 4);
    uint16_t* hnb16  = (uint16_t*)alloc((size_t)N_INST * D_MODEL * 2);
    float*    xz     = (float*)   alloc((size_t)N_INST * 2 * D_INNER * 4);
    float*    xc     = (float*)   alloc((size_t)N_INST * D_INNER * 4);
    uint16_t* xcb16  = (uint16_t*)alloc((size_t)N_INST * D_INNER * 2);
    float*    dbc    = (float*)   alloc((size_t)N_INST * 64 * 4);
    uint16_t* dbcb16 = (uint16_t*)alloc((size_t)N_INST * 64 * 2);
    float*    dtbuf  = (float*)   alloc((size_t)N_INST * D_INNER * 4);
    uint16_t* yb16   = (uint16_t*)alloc((size_t)N_INST * D_INNER * 2);
    float*    hfin   = (float*)   alloc((size_t)N_INST * D_MODEL * 4);
    uint16_t* hfb16  = (uint16_t*)alloc((size_t)N_INST * D_MODEL * 2);
    float*    tbuf   = (float*)   alloc((size_t)N_INST * 128 * 4);
    float*    avec   = (float*)   alloc((size_t)N_INST * 4);
    float*    wsm    = (float*)   alloc((size_t)N_INST * 4);
    float*    pooled = (float*)   alloc((size_t)D_MODEL * 4);
    // chunked-scan scratch
    float*    hseg   = (float*)   alloc((size_t)N_SEG * D_INNER * D_STATE * 4);
    float*    carry  = (float*)   alloc((size_t)N_SEG * D_INNER * D_STATE * 4);
    float*    sumdt  = (float*)   alloc((size_t)N_SEG * D_INNER * 4);

    auto cvt = [&](const float* s, uint16_t* d, int n) {
        cvt_bf16_kernel<<<(n + 255) / 256, 256, 0, stream>>>(s, d, n);
    };

    // --- stage 0: precision conversion -------------------------------------
    cvt(x,        xb16,     N_INST * IN_DIM);
    cvt(fc1_w,    wb_fc1,   D_MODEL * IN_DIM);
    cvt(in_proj,  wb_inprj, N_LAYERS * 2 * D_INNER * D_MODEL);
    cvt(x_proj,   wb_xprj,  N_LAYERS * 64 * D_INNER);
    cvt(dt_w,     wb_dtw,   N_LAYERS * D_INNER * DT_RANK);
    cvt(out_proj, wb_oprj,  N_LAYERS * D_MODEL * D_INNER);
    cvt(attn_w1,  wb_attn1, 128 * D_MODEL);

    // --- fc1 + ReLU ---------------------------------------------------------
    gemm_wmma_bf16<IN_DIM, IN_DIM, 1><<<gemm_grid(N_INST, D_MODEL), 128, 0, stream>>>(
        xb16, wb_fc1, fc1_b, nullptr, hbuf, nullptr, N_INST, D_MODEL);

    // --- mamba layers -------------------------------------------------------
    for (int i = 0; i < N_LAYERS; ++i) {
        ln_kernel<<<N_INST / 8, 256, 0, stream>>>(
            hbuf, ln_w + i * D_MODEL, ln_b + i * D_MODEL, nullptr, hnb16);

        gemm_wmma_bf16<D_MODEL, D_MODEL, 0><<<gemm_grid(N_INST, 2 * D_INNER), 128, 0, stream>>>(
            hnb16, wb_inprj + (size_t)i * 2 * D_INNER * D_MODEL,
            nullptr, nullptr, xz, nullptr, N_INST, 2 * D_INNER);

        conv_silu_kernel<<<(N_INST * D_INNER) / 256, 256, 0, stream>>>(
            xz, conv_w + (size_t)i * D_INNER * D_CONV, conv_b + i * D_INNER,
            xc, xcb16);

        gemm_wmma_bf16<D_INNER, D_INNER, 0><<<gemm_grid(N_INST, 64), 128, 0, stream>>>(
            xcb16, wb_xprj + (size_t)i * 64 * D_INNER,
            nullptr, nullptr, dbc, dbcb16, N_INST, 64);

        gemm_wmma_bf16<DT_RANK, 64, 3><<<gemm_grid(N_INST, D_INNER), 128, 0, stream>>>(
            dbcb16, wb_dtw + (size_t)i * D_INNER * DT_RANK,
            dt_b + i * D_INNER, nullptr, dtbuf, nullptr, N_INST, D_INNER);

        // chunked selective scan (phases A/B/C)
        {
            dim3 gseg(D_INNER / 256, N_SEG);
            scan_phase_a<<<gseg, 256, 0, stream>>>(
                dtbuf, xc, dbc, A_log + (size_t)i * D_INNER * D_STATE,
                hseg, sumdt);
            scan_phase_b<<<(D_INNER * D_STATE) / 256, 256, 0, stream>>>(
                hseg, sumdt, A_log + (size_t)i * D_INNER * D_STATE, carry);
            scan_phase_c<<<gseg, 256, 0, stream>>>(
                dtbuf, xc, dbc, xz, A_log + (size_t)i * D_INNER * D_STATE,
                D_p + i * D_INNER, carry, yb16);
        }

        gemm_wmma_bf16<D_INNER, D_INNER, 0><<<gemm_grid(N_INST, D_MODEL), 128, 0, stream>>>(
            yb16, wb_oprj + (size_t)i * D_MODEL * D_INNER,
            nullptr, hbuf, hbuf, nullptr, N_INST, D_MODEL);
    }

    // --- final LN -----------------------------------------------------------
    ln_kernel<<<N_INST / 8, 256, 0, stream>>>(hbuf, norm_w, norm_b, hfin, hfb16);

    // --- attention pooling --------------------------------------------------
    gemm_wmma_bf16<D_MODEL, D_MODEL, 2><<<gemm_grid(N_INST, 128), 128, 0, stream>>>(
        hfb16, wb_attn1, attn_b1, nullptr, tbuf, nullptr, N_INST, 128);

    attn_score_kernel<<<N_INST / 8, 256, 0, stream>>>(tbuf, attn_w2, attn_b2, avec);
    softmax_kernel<<<1, 1024, 0, stream>>>(avec, wsm);
    pool_kernel<<<1, D_MODEL, 0, stream>>>(wsm, hfin, pooled);
    logits_kernel<<<1, 64, 0, stream>>>(pooled, cls_w, cls_b, out);

    (void)in_sizes; (void)n_in; (void)out_size; (void)ws_size;
}
